// SS2D_84825604096658
// MI455X (gfx1250) — compile-verified, matrix operand order checked by reading, not run on hardware
//
#include <hip/hip_runtime.h>
#include <math.h>

// ---------------------------------------------------------------------------
// SS2D (VMamba-style 2D selective scan) for MI455X / gfx1250.
//  - GEMMs on v_wmma_f32_16x16x32_f16 (f16 operands, f32 accumulate).
//  - Weight (B) matrices pre-packed into per-lane WMMA fragment layout (f16),
//    so the GEMM inner loop is branch-free: 4x b128 A-loads + 1x b256 B-load
//    + 1 wmma per K-step, fully unrolled via template<int K>.
//  - Selective scan: 4 batches x 4 dirs x 256 channels = 4096 independent
//    recurrences, one thread each (wave32), full f32.
//  - Deterministic: no atomics; 4 direction buffers summed in the LN kernel.
// ---------------------------------------------------------------------------

typedef __attribute__((ext_vector_type(16))) _Float16 v16h;
typedef __attribute__((ext_vector_type(8)))  float    v8f;

#define BB       4
#define HH       64
#define WWID     64
#define LL       (HH * WWID)            // 4096
#define NTOK     (BB * LL)              // 16384
#define DMODEL   128
#define DINNER   256
#define DSTATE   16
#define DTRANK   8
#define XPN      (DTRANK + 2 * DSTATE)  // 40 valid cols of x_dbl
#define XPNP     48                     // padded row stride (3 n-tiles)
#define LN_EPS   1e-5f

// f16 A/B fragment K-index (CDNA5 ISA §7.12.2, 16-bit A 16x32)
__device__ __forceinline__ int frag_k(int i, int half) {
    return (i & 7) + ((i >> 3) << 4) + (half << 3);
}

// ---- Pack a row-major f32 (K x ldb) weight into f16 WMMA B-fragments -------
// dst layout: [(kt*ntiles + nt)*32 + lane]*16 + i  (32B contiguous per lane)
__global__ void ss2d_pack_b(const float* __restrict__ src, _Float16* __restrict__ dst,
                            int ntiles, int ldb, int nvalid, int total) {
    const int idx = blockIdx.x * blockDim.x + threadIdx.x;
    if (idx >= total) return;
    const int i    = idx & 15;
    const int lane = (idx >> 4) & 31;
    const int tile = idx >> 9;
    const int nt   = tile % ntiles;
    const int kt   = tile / ntiles;
    const int half = lane >> 4;
    const int k    = kt * 32 + frag_k(i, half);
    const int n    = nt * 16 + (lane & 15);
    dst[idx] = (n < nvalid) ? (_Float16)src[(size_t)k * ldb + n] : (_Float16)0.f;
}

// ---- WMMA GEMM: C[M x ldc] = A[M x K] * Bpacked ----------------------------
// One 16x16 output tile per wave; K fully unrolled (K/32 wmma ops).
template <int K, int NTILES>
__launch_bounds__(256)
__global__ void ss2d_wmma_gemm(const float* __restrict__ A,
                               const _Float16* __restrict__ Bp,
                               float* __restrict__ C, int lda, int ldc) {
    const int wave   = threadIdx.x >> 5;
    const int tile   = blockIdx.x * 8 + wave;
    const int m_tile = tile / NTILES;
    const int n_tile = tile - m_tile * NTILES;
    const int lane   = threadIdx.x & 31;
    const int half   = lane >> 4;

    const float*    arow  = A + (size_t)(m_tile * 16 + (lane & 15)) * lda + (half << 3);
    const _Float16* bbase = Bp + ((size_t)n_tile * 32 + lane) * 16;

    v8f acc = {};
#pragma unroll
    for (int kt = 0; kt < K / 32; ++kt) {
        const float* ap = arow + kt * 32;
        const float4 q0 = *(const float4*)(ap);
        const float4 q1 = *(const float4*)(ap + 4);
        const float4 q2 = *(const float4*)(ap + 16);
        const float4 q3 = *(const float4*)(ap + 20);
        v16h a;
        a[0]  = (_Float16)q0.x; a[1]  = (_Float16)q0.y;
        a[2]  = (_Float16)q0.z; a[3]  = (_Float16)q0.w;
        a[4]  = (_Float16)q1.x; a[5]  = (_Float16)q1.y;
        a[6]  = (_Float16)q1.z; a[7]  = (_Float16)q1.w;
        a[8]  = (_Float16)q2.x; a[9]  = (_Float16)q2.y;
        a[10] = (_Float16)q2.z; a[11] = (_Float16)q2.w;
        a[12] = (_Float16)q3.x; a[13] = (_Float16)q3.y;
        a[14] = (_Float16)q3.z; a[15] = (_Float16)q3.w;

        const v16h b = *(const v16h*)(bbase + (size_t)kt * NTILES * 512);
        acc = __builtin_amdgcn_wmma_f32_16x16x32_f16(
            false, a, false, b, (short)0, acc, false, false);
    }

    const int n = n_tile * 16 + (lane & 15);
    float* crow = C + (size_t)(m_tile * 16 + 8 * half) * ldc + n;
#pragma unroll
    for (int v = 0; v < 8; ++v) crow[(size_t)v * ldc] = acc[v];
}

// ---- Depthwise 3x3 conv (SAME, pad 1) + bias + SiLU ------------------------
__global__ void ss2d_conv_dw_silu(const float* __restrict__ xz,
                                  const float* __restrict__ cw,
                                  const float* __restrict__ cb,
                                  float* __restrict__ xa) {
    const int idx = blockIdx.x * blockDim.x + threadIdx.x;  // NTOK*DINNER
    const int d = idx & (DINNER - 1);
    const int t = idx >> 8;
    const int b = t >> 12;
    const int l = t & (LL - 1);
    const int h = l >> 6;
    const int w = l & 63;
    float acc = cb[d];
#pragma unroll
    for (int kh = 0; kh < 3; ++kh) {
#pragma unroll
        for (int kw = 0; kw < 3; ++kw) {
            const int hy = h + kh - 1;
            const int wx = w + kw - 1;
            if (hy >= 0 && hy < HH && wx >= 0 && wx < WWID) {
                const size_t src = ((size_t)(b * LL + hy * WWID + wx)) * 512 + d;
                acc += xz[src] * cw[(kh * 3 + kw) * DINNER + d];
            }
        }
    }
    xa[idx] = acc / (1.f + __expf(-acc));   // SiLU
}

// ---- Selective scan: one thread per (batch, direction, channel) ------------
__global__ void ss2d_scan(const float* __restrict__ xa,
                          const float* __restrict__ xdbl,
                          const float* __restrict__ w_dt,
                          const float* __restrict__ b_dt,
                          const float* __restrict__ A_log,
                          const float* __restrict__ D_param,
                          float* __restrict__ ys4) {
    const int blk = blockIdx.x;
    const int dir = blk >> 4;
    const int b   = (blk >> 2) & 3;
    const int d   = (blk & 3) * 64 + threadIdx.x;

    float A[DSTATE];
#pragma unroll
    for (int n = 0; n < DSTATE; ++n) A[n] = -__expf(A_log[d * DSTATE + n]);
    float wdt[DTRANK];
#pragma unroll
    for (int r = 0; r < DTRANK; ++r) wdt[r] = w_dt[r * DINNER + d];
    const float dtb = b_dt[d];
    const float Dp  = D_param[d];

    float h[DSTATE];
#pragma unroll
    for (int n = 0; n < DSTATE; ++n) h[n] = 0.f;

    const float* xab = xa   + (size_t)b * LL * DINNER;
    const float* xdb = xdbl + (size_t)b * LL * XPNP;
    float*       yo  = ys4  + (size_t)dir * NTOK * DINNER + (size_t)b * LL * DINNER;

    for (int i = 0; i < LL; ++i) {
        // direction permutation: flip for dirs 2/3, H<->W transpose for 1/3
        const int j = (dir & 2) ? (LL - 1 - i) : i;
        const int s = (dir & 1) ? ((j & 63) * WWID + (j >> 6)) : j;

        const float u = xab[(size_t)s * DINNER + d];
        const float* rp = xdb + (size_t)i * XPNP;   // dt/B/C use row-major i (rep4)
        if (i + 1 < LL) __builtin_prefetch(xdb + (size_t)(i + 1) * XPNP, 0, 1);

        float row[40];
#pragma unroll
        for (int q = 0; q < 10; ++q) {
            const float4 f = *(const float4*)(rp + 4 * q);
            row[4 * q + 0] = f.x; row[4 * q + 1] = f.y;
            row[4 * q + 2] = f.z; row[4 * q + 3] = f.w;
        }

        float dtr = dtb;
#pragma unroll
        for (int r = 0; r < DTRANK; ++r) dtr += wdt[r] * row[r];
        const float dt = (dtr > 20.f) ? dtr : log1pf(__expf(dtr));  // softplus
        const float du = dt * u;

        float y = 0.f;
#pragma unroll
        for (int n = 0; n < DSTATE; ++n) {
            h[n] = h[n] * __expf(dt * A[n]) + du * row[DTRANK + n];
            y += h[n] * row[DTRANK + DSTATE + n];
        }
        yo[(size_t)s * DINNER + d] = y + u * Dp;   // write-back at source location
    }
}

// ---- Sum 4 directions + LayerNorm + SiLU(z) gate ---------------------------
__global__ void ss2d_ln_gate(const float* __restrict__ ys4,
                             const float* __restrict__ xz,
                             const float* __restrict__ ln_w,
                             const float* __restrict__ ln_b,
                             float* __restrict__ yg) {
    const int t = blockIdx.x;
    const int d = threadIdx.x;
    __shared__ float sbuf[8];

    float y = 0.f;
#pragma unroll
    for (int dir = 0; dir < 4; ++dir)
        y += ys4[(size_t)dir * NTOK * DINNER + (size_t)t * DINNER + d];

    float v = y;
#pragma unroll
    for (int o = 16; o > 0; o >>= 1) v += __shfl_down(v, o, 32);
    if ((threadIdx.x & 31) == 0) sbuf[threadIdx.x >> 5] = v;
    __syncthreads();
    float mu = 0.f;
#pragma unroll
    for (int i = 0; i < 8; ++i) mu += sbuf[i];
    mu *= (1.f / DINNER);
    __syncthreads();

    const float c = y - mu;
    v = c * c;
#pragma unroll
    for (int o = 16; o > 0; o >>= 1) v += __shfl_down(v, o, 32);
    if ((threadIdx.x & 31) == 0) sbuf[threadIdx.x >> 5] = v;
    __syncthreads();
    float var = 0.f;
#pragma unroll
    for (int i = 0; i < 8; ++i) var += sbuf[i];
    var *= (1.f / DINNER);

    const float yn = c * rsqrtf(var + LN_EPS) * ln_w[d] + ln_b[d];
    const float z  = xz[(size_t)t * 512 + DINNER + d];
    yg[(size_t)t * DINNER + d] = yn * (z / (1.f + __expf(-z)));
}

// ---------------------------------------------------------------------------
extern "C" void kernel_launch(void* const* d_in, const int* in_sizes, int n_in,
                              void* d_out, int out_size, void* d_ws, size_t ws_size,
                              hipStream_t stream) {
    const float* x       = (const float*)d_in[0];
    const float* w_in    = (const float*)d_in[1];
    const float* conv_w  = (const float*)d_in[2];
    const float* conv_b  = (const float*)d_in[3];
    const float* w_xproj = (const float*)d_in[4];
    const float* w_dt    = (const float*)d_in[5];
    const float* b_dt    = (const float*)d_in[6];
    const float* A_log   = (const float*)d_in[7];
    const float* D_param = (const float*)d_in[8];
    const float* ln_w    = (const float*)d_in[9];
    const float* ln_b    = (const float*)d_in[10];
    const float* w_out   = (const float*)d_in[11];
    float* out = (float*)d_out;

    float* ws   = (float*)d_ws;
    float* xz   = ws;                               // NTOK*512
    float* xa   = xz   + (size_t)NTOK * 512;        // NTOK*256 (reused as yg)
    float* xdbl = xa   + (size_t)NTOK * DINNER;     // NTOK*48 (padded, cols 40-47 = 0)
    float* ys4  = xdbl + (size_t)NTOK * XPNP;       // 4*NTOK*256
    _Float16* winp  = (_Float16*)(ys4 + (size_t)4 * NTOK * DINNER);  // 128*512 halves
    _Float16* xprjp = winp  + (size_t)DMODEL * 512;                  // 256*48 halves
    _Float16* woutp = xprjp + (size_t)DINNER * XPNP;                 // 256*128 halves

    // 0) pack weights into f16 WMMA B-fragments
    ss2d_pack_b<<<(DMODEL * 512) / 256, 256, 0, stream>>>(w_in, winp, 32, 512, 512, DMODEL * 512);
    ss2d_pack_b<<<(DINNER * XPNP) / 256, 256, 0, stream>>>(w_xproj, xprjp, 3, XPN, XPN, DINNER * XPNP);
    ss2d_pack_b<<<(DINNER * DMODEL) / 256, 256, 0, stream>>>(w_out, woutp, 8, DMODEL, DMODEL, DINNER * DMODEL);

    // 1) in_proj: (16384 x 128) * (128 x 512) -> xz
    ss2d_wmma_gemm<DMODEL, 32><<<4096, 256, 0, stream>>>(x, winp, xz, DMODEL, 512);

    // 2) depthwise 3x3 conv + SiLU -> xa
    ss2d_conv_dw_silu<<<(NTOK * DINNER) / 256, 256, 0, stream>>>(xz, conv_w, conv_b, xa);

    // 3) x_proj: (16384 x 256) * (256 x 48pad) -> xdbl
    ss2d_wmma_gemm<DINNER, 3><<<384, 256, 0, stream>>>(xa, xprjp, xdbl, DINNER, XPNP);

    // 4) selective scan, 4 dirs x 4 batches x 4 channel-chunks
    ss2d_scan<<<64, 64, 0, stream>>>(xa, xdbl, w_dt, b_dt, A_log, D_param, ys4);

    // 5) sum dirs + LN + gate -> yg (reuses xa region)
    ss2d_ln_gate<<<NTOK, 256, 0, stream>>>(ys4, xz, ln_w, ln_b, xa);

    // 6) out_proj: (16384 x 256) * (256 x 128) -> out
    ss2d_wmma_gemm<DINNER, 8><<<1024, 256, 0, stream>>>(xa, woutp, out, DINNER, DMODEL);
}
